// Model_21706764714209
// MI455X (gfx1250) — matrix operand order repaired and checked by reading, not compiled, not run on hardware
//
#include <hip/hip_runtime.h>
#include <hip/hip_bf16.h>
#include <math.h>

typedef __attribute__((ext_vector_type(8)))  __bf16 bf16x8;
typedef __attribute__((ext_vector_type(16))) __bf16 bf16x16;
typedef __attribute__((ext_vector_type(8)))  float  f32x8;
typedef __attribute__((ext_vector_type(4)))  int    v4i;

#define SEQLEN   720
#define PREDP    720
#define CHN      862
#define BATCHB   64
#define NDIM     360            // IN_LEN = K of the GEMM
#define KPAD     384            // padded to multiple of 32 (bf16 WMMA K)
#define ROWS     (BATCHB*CHN)   // 55168 = 3448 * 16
#define PLANE    (PREDP*CHN)    // 620640

#if __has_builtin(__builtin_amdgcn_global_load_async_to_lds_b128) && \
    __has_builtin(__builtin_amdgcn_s_wait_asynccnt)
#define HAVE_ASYNC_LDS 1
#else
#define HAVE_ASYNC_LDS 0
#endif

// ---------------------------------------------------------------------------
// Kernel 0: E = (1/N) * W_lo @ D  (720 x 360, padded to 384), bf16 hi/lo split.
// D[k,n] = cos(pi*(n+0.5)*k/N)*sqrt(2/N), row k=0 scaled by 1/sqrt(2).
// Cosine series via the stable 3-term recurrence (2 FMA per k, no cosf in the
// inner loop). Also emits rsb[p] = {sum_n W_lo[p,n], b_lo[p]}.
// ---------------------------------------------------------------------------
__global__ void build_E(const float* __restrict__ Wlo, const float* __restrict__ blo,
                        __bf16* __restrict__ Eh, __bf16* __restrict__ El,
                        float2* __restrict__ rsb) {
    __shared__ float wrow[NDIM];
    __shared__ float red[256];
    const int p = blockIdx.x;
    const float* w = Wlo + p * NDIM;

    float part = 0.0f;
    for (int k = threadIdx.x; k < NDIM; k += 256) { float v = w[k]; wrow[k] = v; part += v; }
    red[threadIdx.x] = part;
    __syncthreads();
    if (threadIdx.x == 0) {
        float s = 0.0f;
        for (int i = 0; i < 256; ++i) s += red[i];
        rsb[p] = make_float2(s, blo[p]);
    }

    const float scale     = sqrtf(2.0f / (float)NDIM) / (float)NDIM;
    const float inv_sqrt2 = 0.70710678118654752f;
    for (int n = threadIdx.x; n < KPAD; n += 256) {
        float e = 0.0f;
        if (n < NDIM) {
            float th    = 3.14159265358979323846f * ((float)n + 0.5f) / (float)NDIM;
            float c1    = cosf(th);
            float two_c = 2.0f * c1;
            float cp = 1.0f, cc = c1;
            float acc = wrow[0] * inv_sqrt2;          // k = 0 row has 1/sqrt(2)
            for (int k = 1; k < NDIM; ++k) {
                acc = fmaf(wrow[k], cc, acc);
                float cn = fmaf(two_c, cc, -cp);
                cp = cc; cc = cn;
            }
            e = acc * scale;
        }
        __bf16 h = (__bf16)e;
        Eh[p * KPAD + n] = h;
        El[p * KPAD + n] = (__bf16)(e - (float)h);
    }
}

// ---------------------------------------------------------------------------
// Kernel 1: per-(b,c) mean, low-pass pair-sum, bf16 hi/lo split of lp,
// and per-row fused params prm[r] = {s, t, mean, bits(b*PLANE + c)}.
// x is streamed with non-temporal loads (read-once) so the lp arrays we write
// stay resident in the 192MB L2 for the GEMM's A-matrix reads.
// ---------------------------------------------------------------------------
__global__ void preprocess(const float* __restrict__ x,
                           const float* __restrict__ w1, const float* __restrict__ b1,
                           const float* __restrict__ w2, const float* __restrict__ b2,
                           __bf16* __restrict__ lph, __bf16* __restrict__ lpl,
                           float4* __restrict__ prm) {
    __shared__ float xs[16 * 721];
    __shared__ float ps[256];
    __shared__ float meanS[16];

    const int b  = blockIdx.y;
    const int c0 = blockIdx.x * 16;
    const float* xb = x + (size_t)b * PLANE;

    // load 720 x 16 tile (coalesced 64B rows), zero-fill invalid channels
    for (int idx = threadIdx.x; idx < SEQLEN * 16; idx += 256) {
        int cl = idx & 15, l = idx >> 4;
        int c  = c0 + cl;
        float v = (c < CHN) ? __builtin_nontemporal_load(&xb[(size_t)l * CHN + c]) : 0.0f;
        xs[cl * 721 + l] = v;
    }
    __syncthreads();

    // partial sums: 16 parts of 45 per channel
    {
        int cl = threadIdx.x & 15, part = threadIdx.x >> 4;
        const float* col = xs + cl * 721 + part * 45;
        float s = 0.0f;
        #pragma unroll
        for (int j = 0; j < 45; ++j) s += col[j];
        ps[threadIdx.x] = s;
    }
    __syncthreads();

    if (threadIdx.x < 16) {
        int cl = threadIdx.x;
        float s = 0.0f;
        for (int j = 0; j < 16; ++j) s += ps[j * 16 + cl];
        float mean = s * (1.0f / 720.0f);
        meanS[cl] = mean;
        int c = c0 + cl;
        if (c < CHN) {
            float W1 = w1[c], B1 = b1[c], W2 = w2[c], B2 = b2[c];
            float sc = fmaf(W1, W2, 1.0f);     // 1 + w1*w2
            float tt = fmaf(W1, B2, B1);       // w1*b2 + b1
            int r = b * CHN + c;
            prm[r] = make_float4(sc, tt, mean, __int_as_float(b * PLANE + c));
        }
    }
    __syncthreads();

    // lp = (x[2n]+x[2n+1]-2*mean)/sqrt(2), split to bf16 hi/lo, K-padded
    for (int idx = threadIdx.x; idx < 16 * KPAD; idx += 256) {
        int cl = idx / KPAD;
        int n  = idx - cl * KPAD;      // consecutive threads -> consecutive n (coalesced)
        int c  = c0 + cl;
        if (c >= CHN) continue;
        int r = b * CHN + c;
        float v = 0.0f;
        if (n < NDIM) {
            float a0 = xs[cl * 721 + 2 * n];
            float a1 = xs[cl * 721 + 2 * n + 1];
            v = (a0 + a1 - 2.0f * meanS[cl]) * 0.70710678118654752f;
        }
        __bf16 h = (__bf16)v;
        lph[(size_t)r * KPAD + n] = h;
        lpl[(size_t)r * KPAD + n] = (__bf16)(v - (float)h);
    }
}

// ---------------------------------------------------------------------------
// B-tile staging: 16 rows x KPAD (hi+lo) = 2 x 12288B, 16B chunks per lane.
// Async path uses GLOBAL_LOAD_ASYNC_TO_LDS_B128 (ASYNCcnt, no VGPR roundtrip).
// ---------------------------------------------------------------------------
__device__ __forceinline__ void stage_tile(const __bf16* __restrict__ Eh,
                                           const __bf16* __restrict__ El,
                                           __bf16* sBh, __bf16* sBl,
                                           int nt, int tid) {
    const uint4* gh = (const uint4*)(Eh + (size_t)nt * 16 * KPAD);
    const uint4* gl = (const uint4*)(El + (size_t)nt * 16 * KPAD);
    uint4* dh = (uint4*)sBh;
    uint4* dl = (uint4*)sBl;
#if HAVE_ASYNC_LDS
    #pragma unroll
    for (int i = 0; i < 3; ++i) {
        __builtin_amdgcn_global_load_async_to_lds_b128(
            (__attribute__((address_space(1))) v4i*)(uintptr_t)(gh + tid + i * 256),
            (__attribute__((address_space(3))) v4i*)(uintptr_t)(dh + tid + i * 256), 0, 0);
        __builtin_amdgcn_global_load_async_to_lds_b128(
            (__attribute__((address_space(1))) v4i*)(uintptr_t)(gl + tid + i * 256),
            (__attribute__((address_space(3))) v4i*)(uintptr_t)(dl + tid + i * 256), 0, 0);
    }
#else
    #pragma unroll
    for (int i = 0; i < 3; ++i) {
        dh[tid + i * 256] = gh[tid + i * 256];
        dl[tid + i * 256] = gl[tid + i * 256];
    }
#endif
}

// ---------------------------------------------------------------------------
// Kernel 2: out[r,p] = s[r]*(lp[r,:]·E[p,:]) + t[r]*rs[p] + b_lo[p] + mean[r]
// stored transposed as out[b,p,c]. Split-bf16 WMMA (AhBh + AhBl + AlBh).
// 8 waves/block; each wave: one 16-row tile, all 12 K-step A frags in VGPRs,
// E tiles double-buffered in LDS with async prefetch of tile nt+1 overlapping
// the wmma chain of tile nt.
// ---------------------------------------------------------------------------
__global__ __launch_bounds__(256) void gemm_out(
    const __bf16* __restrict__ lph, const __bf16* __restrict__ lpl,
    const __bf16* __restrict__ Eh,  const __bf16* __restrict__ El,
    const float2* __restrict__ rsb, const float4* __restrict__ prm,
    float* __restrict__ out) {

    __shared__ __bf16 sBh[2][16 * KPAD];
    __shared__ __bf16 sBl[2][16 * KPAD];
    __shared__ float  tile[8 * 320];   // per-wave 16x16 transpose staging, stride 20

    const int tid    = threadIdx.x;
    const int wave   = tid >> 5;
    const int lane   = tid & 31;
    const int lane15 = lane & 15;
    const bool hi    = lane >= 16;
    const int r0     = (blockIdx.x * 8 + wave) * 16;

    // ---- A fragments (hi/lo) for all 12 K-steps, per ISA 16-bit A layout:
    // lanes 0-15: M=lane, K in {kb+0..7, kb+16..23}; lanes 16-31: {kb+8..15, kb+24..31}
    bf16x16 Ah[12], Al[12];
    {
        const __bf16* bh = lph + (size_t)(r0 + lane15) * KPAD + (hi ? 8 : 0);
        const __bf16* bl = lpl + (size_t)(r0 + lane15) * KPAD + (hi ? 8 : 0);
        #pragma unroll
        for (int ks = 0; ks < 12; ++ks) {
            int kb = ks * 32;
            bf16x8 a0 = *(const bf16x8*)(bh + kb);
            bf16x8 a1 = *(const bf16x8*)(bh + kb + 16);
            Ah[ks] = __builtin_shufflevector(a0, a1, 0,1,2,3,4,5,6,7,8,9,10,11,12,13,14,15);
            bf16x8 l0 = *(const bf16x8*)(bl + kb);
            bf16x8 l1 = *(const bf16x8*)(bl + kb + 16);
            Al[ks] = __builtin_shufflevector(l0, l1, 0,1,2,3,4,5,6,7,8,9,10,11,12,13,14,15);
        }
    }
    const float4 myprm = prm[r0 + lane15];
    const int rowoff   = __float_as_int(myprm.w);
    float* mytile      = tile + wave * 320;

#if HAVE_ASYNC_LDS
    stage_tile(Eh, El, sBh[0], sBl[0], 0, tid);   // prefetch tile 0
#endif

    for (int nt = 0; nt < 45; ++nt) {
#if HAVE_ASYNC_LDS
        const int cb = nt & 1;
        __builtin_amdgcn_s_wait_asynccnt(0);      // this wave's copies for tile nt done
        __syncthreads();                          // all waves' copies visible; prev reads done
        if (nt + 1 < 45)                          // prefetch nt+1 overlapped with compute
            stage_tile(Eh, El, sBh[cb ^ 1], sBl[cb ^ 1], nt + 1, tid);
#else
        const int cb = 0;
        __syncthreads();
        stage_tile(Eh, El, sBh[0], sBl[0], nt, tid);
        __syncthreads();
#endif

        f32x8 acc = {0.f,0.f,0.f,0.f,0.f,0.f,0.f,0.f};
        // B layout: lane = N column, lanes 0-15 hold K {kb+0..7, kb+16..23}, lanes 16-31 hold {kb+8..15, kb+24..31}
        const __bf16* pbh = sBh[cb] + lane15 * KPAD + (hi ? 8 : 0);
        const __bf16* pbl = sBl[cb] + lane15 * KPAD + (hi ? 8 : 0);
        #pragma unroll
        for (int ks = 0; ks < 12; ++ks) {
            int kb = ks * 32;
            bf16x8 b0 = *(const bf16x8*)(pbh + kb);
            bf16x8 b1 = *(const bf16x8*)(pbh + kb + 16);
            bf16x16 Bh = __builtin_shufflevector(b0, b1, 0,1,2,3,4,5,6,7,8,9,10,11,12,13,14,15);
            bf16x8 c0 = *(const bf16x8*)(pbl + kb);
            bf16x8 c1 = *(const bf16x8*)(pbl + kb + 16);
            bf16x16 Bl = __builtin_shufflevector(c0, c1, 0,1,2,3,4,5,6,7,8,9,10,11,12,13,14,15);
            acc = __builtin_amdgcn_wmma_f32_16x16x32_bf16(false, Ah[ks], false, Bh, (short)0, acc, false, false);
            acc = __builtin_amdgcn_wmma_f32_16x16x32_bf16(false, Ah[ks], false, Bl, (short)0, acc, false, false);
            acc = __builtin_amdgcn_wmma_f32_16x16x32_bf16(false, Al[ks], false, Bh, (short)0, acc, false, false);
        }

        // ---- epilogue: affine + transpose-through-LDS + coalesced (over c) stores
        const int p0 = nt * 16;
        const float2 rb = rsb[p0 + lane15];
        #pragma unroll
        for (int i = 0; i < 8; ++i) {
            int m   = i + (hi ? 8 : 0);               // C/D layout: lanes>=16 hold M=i+8
            float s  = __shfl(myprm.x, m, 32);
            float t  = __shfl(myprm.y, m, 32);
            float mn = __shfl(myprm.z, m, 32);
            float val = fmaf(s, acc[i], fmaf(t, rb.x, rb.y + mn));
            mytile[m * 20 + lane15] = val;            // stride 20 -> conflict-free
        }
        __builtin_amdgcn_wave_barrier();              // DS ops are in-order per wave
        #pragma unroll
        for (int j = 0; j < 8; ++j) {
            int p = p0 + 2 * j + (hi ? 1 : 0);
            float val = mytile[lane15 * 20 + 2 * j + (hi ? 1 : 0)];
            // streamed output (read by nobody) -> non-temporal; contiguous over 16 c's
            __builtin_nontemporal_store(val, &out[(size_t)rowoff + (size_t)p * CHN]);
        }
        __builtin_amdgcn_wave_barrier();
    }
}

// ---------------------------------------------------------------------------
extern "C" void kernel_launch(void* const* d_in, const int* in_sizes, int n_in,
                              void* d_out, int out_size, void* d_ws, size_t ws_size,
                              hipStream_t stream) {
    const float* x   = (const float*)d_in[0];
    const float* w1  = (const float*)d_in[1];
    const float* b1  = (const float*)d_in[2];
    const float* w2  = (const float*)d_in[3];
    const float* b2  = (const float*)d_in[4];
    const float* Wlo = (const float*)d_in[5];
    const float* blo = (const float*)d_in[6];
    float* out = (float*)d_out;

    char* ws = (char*)d_ws;
    size_t off = 0;
    auto take = [&](size_t bytes) -> void* {
        void* p = ws + off;
        off = (off + bytes + 255) & ~(size_t)255;
        return p;
    };
    __bf16* Eh  = (__bf16*)take((size_t)PREDP * KPAD * 2);
    __bf16* El  = (__bf16*)take((size_t)PREDP * KPAD * 2);
    float2* rsb = (float2*)take((size_t)PREDP * sizeof(float2));
    float4* prm = (float4*)take((size_t)ROWS * sizeof(float4));
    __bf16* lph = (__bf16*)take((size_t)ROWS * KPAD * 2);
    __bf16* lpl = (__bf16*)take((size_t)ROWS * KPAD * 2);

    build_E<<<PREDP, 256, 0, stream>>>(Wlo, blo, Eh, El, rsb);
    preprocess<<<dim3((CHN + 15) / 16, BATCHB), 256, 0, stream>>>(
        x, w1, b1, w2, b2, lph, lpl, prm);
    gemm_out<<<ROWS / (16 * 8), 256, 0, stream>>>(lph, lpl, Eh, El, rsb, prm, out);
}
